// GAT_40089224741167
// MI455X (gfx1250) — matrix operand order repaired and checked by reading, not compiled, not run on hardware
//
#include <hip/hip_runtime.h>
#include <hip/hip_bf16.h>

// ---------------------------------------------------------------------------
// GAT 2-layer forward for MI455X (gfx1250, wave32).
// GEMMs use v_wmma_f32_16x16x32_bf16 with software-pipelined global->LDS
// staging; graph aggregation uses fp32 L2 atomics (acc fits in 192MB L2).
// ---------------------------------------------------------------------------

typedef __bf16 bf16;
typedef __attribute__((ext_vector_type(4)))  __bf16 v4bf;
typedef __attribute__((ext_vector_type(8)))  __bf16 v8bf;
typedef __attribute__((ext_vector_type(16))) __bf16 v16bf;
typedef __attribute__((ext_vector_type(8)))  float  v8f;

__device__ __forceinline__ v16bf join16(v8bf lo, v8bf hi) {
    union { v8bf h[2]; v16bf f; } u;
    u.h[0] = lo; u.h[1] = hi;
    return u.f;
}

// ---------------------------------------------------------------------------
// WMMA GEMM: C[M,Nmat] = A[M,K] @ B[K,Nmat], fp32 in/out, bf16 MACs.
// Block tile 128x64, 8 waves = 4(M) x 2(N), each wave 2x2 wmma tiles, K step 32.
// Software pipelined: next K-tile global loads overlap current tile's WMMA.
// ---------------------------------------------------------------------------
#define TM 128
#define TN 64
#define TK 32
#define LDP (TK + 8)   // padded LDS row stride (40 elems = 80B, 16B aligned)

__global__ __launch_bounds__(256, 2) void wmma_gemm_bf16(
    const float* __restrict__ A, const float* __restrict__ B,
    float* __restrict__ C, int M, int Nmat, int K)
{
    __shared__ __bf16 Al[TM][LDP];
    __shared__ __bf16 Bl[TN][LDP];   // stored transposed: [n][k]

    const int tid  = threadIdx.x;
    const int lane = tid & 31;
    const int wave = tid >> 5;
    const int wm   = wave & 3;          // 0..3  (M direction)
    const int wn   = wave >> 2;         // 0..1  (N direction)
    const int blockM = blockIdx.x * TM;
    const int blockN = blockIdx.y * TN;
    const int halfsel = lane >> 4;      // 0: lanes 0-15, 1: lanes 16-31
    const int l15 = lane & 15;

    // ---- fixed per-thread staging pointers (advance by TK each iteration)
    const float* aP[4]; __bf16* aD[4];
    #pragma unroll
    for (int t = 0; t < 4; ++t) {
        int idx = tid + t * 256;            // 0..1023 float4 slots of 128x32 tile
        int r = idx >> 3, c4 = idx & 7;
        int row = blockM + r; if (row >= M) row = M - 1;   // clamp (stores guarded)
        aP[t] = A + (size_t)row * K + (c4 << 2);
        aD[t] = &Al[r][c4 << 2];
    }
    const float* bP[8]; __bf16* bD[8];
    #pragma unroll
    for (int t = 0; t < 8; ++t) {
        int idx = tid + t * 256;            // 0..2047 elems of 32x64 tile
        int kr = idx >> 6, nc = idx & 63;   // coalesced along nc
        bP[t] = B + (size_t)kr * Nmat + blockN + nc;
        bD[t] = &Bl[nc][kr];
    }

    // ---- prologue: load K-tile 0 into registers
    float4 aR[4]; float bR[8];
    #pragma unroll
    for (int t = 0; t < 4; ++t) aR[t] = *(const float4*)aP[t];
    #pragma unroll
    for (int t = 0; t < 8; ++t) bR[t] = *bP[t];

    v8f acc[2][2] = {};
    const int ak0 = halfsel ? 8  : 0;   // A: lanes>=16 hold K 8-15 / 24-31
    const int bk0 = halfsel ? 8  : 0;   // B: lanes>=16 hold K 8-15 / 24-31
    const __bf16* pa0 = &Al[wm * 32 + l15][0];
    const __bf16* pa1 = &Al[wm * 32 + 16 + l15][0];
    const __bf16* pb0 = &Bl[wn * 32 + l15][0];
    const __bf16* pb1 = &Bl[wn * 32 + 16 + l15][0];

    for (int kk = 0; kk < K; kk += TK) {
        __syncthreads();                    // prior WMMA reads of LDS done
        // ---- commit register tile -> LDS (fp32 -> bf16)
        #pragma unroll
        for (int t = 0; t < 4; ++t) {
            v4bf b4;
            b4[0] = (bf16)aR[t].x; b4[1] = (bf16)aR[t].y;
            b4[2] = (bf16)aR[t].z; b4[3] = (bf16)aR[t].w;
            *(v4bf*)aD[t] = b4;
        }
        #pragma unroll
        for (int t = 0; t < 8; ++t) *bD[t] = (bf16)bR[t];
        __syncthreads();

        // ---- issue next tile's global loads (overlap with WMMA below)
        if (kk + TK < K) {
            #pragma unroll
            for (int t = 0; t < 4; ++t) { aP[t] += TK; aR[t] = *(const float4*)aP[t]; }
            #pragma unroll
            for (int t = 0; t < 8; ++t) { bP[t] += (size_t)TK * Nmat; bR[t] = *bP[t]; }
        }

        // ---- fragments (wave32 layout per CDNA5 ISA 7.12.2) + 4 WMMAs
        v16bf af0 = join16(*(const v8bf*)(pa0 + ak0), *(const v8bf*)(pa0 + ak0 + 16));
        v16bf af1 = join16(*(const v8bf*)(pa1 + ak0), *(const v8bf*)(pa1 + ak0 + 16));
        v16bf bf0 = join16(*(const v8bf*)(pb0 + bk0), *(const v8bf*)(pb0 + bk0 + 16));
        v16bf bf1 = join16(*(const v8bf*)(pb1 + bk0), *(const v8bf*)(pb1 + bk0 + 16));
        acc[0][0] = __builtin_amdgcn_wmma_f32_16x16x32_bf16(false, af0, false, bf0, (short)0, acc[0][0], false, false);
        acc[0][1] = __builtin_amdgcn_wmma_f32_16x16x32_bf16(false, af0, false, bf1, (short)0, acc[0][1], false, false);
        acc[1][0] = __builtin_amdgcn_wmma_f32_16x16x32_bf16(false, af1, false, bf0, (short)0, acc[1][0], false, false);
        acc[1][1] = __builtin_amdgcn_wmma_f32_16x16x32_bf16(false, af1, false, bf1, (short)0, acc[1][1], false, false);
    }

    // ---- epilogue: fast unguarded path for full tiles (all but last block)
    const bool full = (blockM + TM) <= M;
    #pragma unroll
    for (int i = 0; i < 2; ++i)
        #pragma unroll
        for (int j = 0; j < 2; ++j) {
            int col   = blockN + wn * 32 + j * 16 + l15;
            int mbase = blockM + wm * 32 + i * 16 + halfsel * 8;
            float* cp = C + (size_t)mbase * Nmat + col;
            if (full) {
                #pragma unroll
                for (int r = 0; r < 8; ++r) { *cp = acc[i][j][r]; cp += Nmat; }
            } else {
                #pragma unroll
                for (int r = 0; r < 8; ++r) { if (mbase + r < M) *cp = acc[i][j][r]; cp += Nmat; }
            }
        }
}

// ---------------------------------------------------------------------------
// Per-node attention logits: al_src[n,h] = sum_c h[n,h*C+c]*a_src[h,c], same dst.
// ---------------------------------------------------------------------------
__global__ void att_logits(const float* __restrict__ h,
                           const float* __restrict__ a_src,
                           const float* __restrict__ a_dst,
                           float* __restrict__ o_src, float* __restrict__ o_dst,
                           long long Nn, int H, int C)
{
    long long idx = (long long)blockIdx.x * blockDim.x + threadIdx.x;
    if (idx >= Nn * H) return;
    long long n = idx / H;
    int hh = (int)(idx % H);
    const float* hp = h + (size_t)n * H * C + (size_t)hh * C;
    const float* as = a_src + (size_t)hh * C;
    const float* ad = a_dst + (size_t)hh * C;
    float s1 = 0.f, s2 = 0.f;
    for (int c = 0; c < C; ++c) { float v = hp[c]; s1 += v * as[c]; s2 += v * ad[c]; }
    o_src[idx] = s1; o_dst[idx] = s2;
}

// monotone float<->uint mapping for atomic max on floats (incl. negatives)
__device__ __forceinline__ unsigned f2ord(float f) {
    unsigned u = __float_as_uint(f);
    return (u & 0x80000000u) ? ~u : (u | 0x80000000u);
}
__device__ __forceinline__ float ord2f(unsigned u) {
    unsigned b = (u & 0x80000000u) ? (u & 0x7FFFFFFFu) : ~u;
    return __uint_as_float(b);
}
#define ORD_NEG_INF 0x007FFFFFu   // f2ord(-inf)

__global__ void fill_u32(unsigned* p, unsigned v, long long n) {
    long long i = (long long)blockIdx.x * blockDim.x + threadIdx.x;
    if (i < n) p[i] = v;
}
__global__ void fill_f32(float* p, float v, long long n) {
    long long i = (long long)blockIdx.x * blockDim.x + threadIdx.x;
    if (i < n) p[i] = v;
}

// ---------------------------------------------------------------------------
// Edge pass 1: e = leaky_relu(al_src[src]+al_dst[dst]); segment max via atomics.
// Edges [0,E) come from edge_index; [E, E+N) are self loops.
// ---------------------------------------------------------------------------
__global__ void edge_max(const int* __restrict__ ei,
                         const float* __restrict__ als, const float* __restrict__ ald,
                         float* __restrict__ ebuf, unsigned* __restrict__ mbuf,
                         long long E, long long Nn, int H)
{
    long long e = (long long)blockIdx.x * blockDim.x + threadIdx.x;
    if (e >= E + Nn) return;
    long long s = (e < E) ? (long long)ei[e]     : (e - E);
    long long d = (e < E) ? (long long)ei[E + e] : (e - E);
    for (int hh = 0; hh < H; ++hh) {
        float v = als[s * H + hh] + ald[d * H + hh];
        v = v > 0.f ? v : 0.2f * v;                 // leaky relu
        ebuf[e * H + hh] = v;
        atomicMax(&mbuf[d * H + hh], f2ord(v));
    }
}

// Edge pass 2: ex = exp(e - m[dst]); segment sum of ex.
__global__ void edge_exp(const int* __restrict__ ei,
                         float* __restrict__ ebuf, const unsigned* __restrict__ mbuf,
                         float* __restrict__ denom, long long E, long long Nn, int H)
{
    long long e = (long long)blockIdx.x * blockDim.x + threadIdx.x;
    if (e >= E + Nn) return;
    long long d = (e < E) ? (long long)ei[E + e] : (e - E);
    for (int hh = 0; hh < H; ++hh) {
        float m  = ord2f(mbuf[d * H + hh]);
        float ex = __expf(ebuf[e * H + hh] - m);
        ebuf[e * H + hh] = ex;
        atomicAdd(&denom[d * H + hh], ex);
    }
}

// Edge pass 3: acc[dst] += alpha * h[src], one wave per edge, F = H*C floats.
__global__ void edge_scatter(const int* __restrict__ ei,
                             const float* __restrict__ ebuf, const float* __restrict__ denom,
                             const float* __restrict__ h, float* __restrict__ acc,
                             long long E, long long Nn, int H, int C)
{
    const int lane = threadIdx.x & 31;
    long long e = (long long)blockIdx.x * (blockDim.x >> 5) + (threadIdx.x >> 5);
    if (e >= E + Nn) return;
    long long s = (e < E) ? (long long)ei[e]     : (e - E);
    long long d = (e < E) ? (long long)ei[E + e] : (e - E);
    const int F = H * C;
    for (int i = lane; i < F; i += 32) {
        int hh = i / C;
        float alpha = ebuf[e * H + hh] / (denom[d * H + hh] + 1e-16f);
        atomicAdd(&acc[d * (long long)F + i], alpha * h[s * (long long)F + i]);
    }
}

// Layer-1 finalize: x2 = elu(acc + b), in place.
__global__ void bias_elu(float* __restrict__ acc, const float* __restrict__ b,
                         long long total, int F)
{
    long long i = (long long)blockIdx.x * blockDim.x + threadIdx.x;
    if (i >= total) return;
    float v = acc[i] + b[(int)(i % F)];
    acc[i] = v > 0.f ? v : (__expf(v) - 1.f);
}

// Layer-2 finalize: out = log_softmax(acc + b2), C = 64, one wave per node.
__global__ void bias_logsoftmax(const float* __restrict__ acc, const float* __restrict__ b,
                                float* __restrict__ out, long long Nn)
{
    const int lane = threadIdx.x & 31;
    long long n = (long long)blockIdx.x * (blockDim.x >> 5) + (threadIdx.x >> 5);
    if (n >= Nn) return;
    float v0 = acc[n * 64 + lane]      + b[lane];
    float v1 = acc[n * 64 + 32 + lane] + b[32 + lane];
    float m = fmaxf(v0, v1);
    #pragma unroll
    for (int o = 16; o >= 1; o >>= 1) m = fmaxf(m, __shfl_xor(m, o, 32));
    float s = __expf(v0 - m) + __expf(v1 - m);
    #pragma unroll
    for (int o = 16; o >= 1; o >>= 1) s += __shfl_xor(s, o, 32);
    float lse = m + __logf(s);
    out[n * 64 + lane]      = v0 - lse;
    out[n * 64 + 32 + lane] = v1 - lse;
}

// ---------------------------------------------------------------------------
extern "C" void kernel_launch(void* const* d_in, const int* in_sizes, int n_in,
                              void* d_out, int out_size, void* d_ws, size_t ws_size,
                              hipStream_t stream)
{
    const float* x   = (const float*)d_in[0];
    const int*   ei  = (const int*)  d_in[1];
    const float* W1  = (const float*)d_in[2];
    const float* as1 = (const float*)d_in[3];
    const float* ad1 = (const float*)d_in[4];
    const float* b1  = (const float*)d_in[5];
    const float* W2  = (const float*)d_in[6];
    const float* as2 = (const float*)d_in[7];
    const float* ad2 = (const float*)d_in[8];
    const float* b2  = (const float*)d_in[9];
    float* out = (float*)d_out;

    const long long NN = 50000, FIN = 512, H1 = 4, C1 = 64, F1 = 256, C2 = 64;
    const long long E   = (long long)in_sizes[1] / 2;
    const long long TOT = E + NN;

    // workspace layout (floats); layer-2 buffers reuse h1 region (dead after
    // layer-1 edge_scatter).
    float* h1   = (float*)d_ws;                 // NN*F1
    float* acc1 = h1   + NN * F1;               // NN*F1
    float* e1   = acc1 + NN * F1;               // TOT*H1
    float* als1 = e1   + TOT * H1;              // NN*H1
    float* ald1 = als1 + NN * H1;               // NN*H1
    unsigned* m1 = (unsigned*)(ald1 + NN * H1); // NN*H1
    float* den1 = (float*)m1 + NN * H1;         // NN*H1
    float* h2   = h1;                           // NN*C2   (reuse)
    float* acc2 = h2   + NN * C2;               // NN*C2
    float* e2   = acc2 + NN * C2;               // TOT
    float* als2 = e2   + TOT;                   // NN
    float* ald2 = als2 + NN;                    // NN
    unsigned* m2 = (unsigned*)(ald2 + NN);      // NN
    float* den2 = (float*)m2 + NN;              // NN

    const dim3 blk(256);
    auto cdiv = [](long long a, long long b) { return (unsigned)((a + b - 1) / b); };

    // ===== Layer 1 =====
    dim3 g1(cdiv(NN, 128), (unsigned)(F1 / 64));
    wmma_gemm_bf16<<<g1, blk, 0, stream>>>(x, W1, h1, (int)NN, (int)F1, (int)FIN);

    att_logits<<<cdiv(NN * H1, 256), blk, 0, stream>>>(h1, as1, ad1, als1, ald1, NN, (int)H1, (int)C1);

    fill_u32<<<cdiv(NN * H1, 256), blk, 0, stream>>>(m1, ORD_NEG_INF, NN * H1);
    fill_f32<<<cdiv(NN * H1, 256), blk, 0, stream>>>(den1, 0.f, NN * H1);
    fill_f32<<<cdiv(NN * F1, 256), blk, 0, stream>>>(acc1, 0.f, NN * F1);

    edge_max    <<<cdiv(TOT, 256), blk, 0, stream>>>(ei, als1, ald1, e1, m1, E, NN, (int)H1);
    edge_exp    <<<cdiv(TOT, 256), blk, 0, stream>>>(ei, e1, m1, den1, E, NN, (int)H1);
    edge_scatter<<<cdiv(TOT, 8),   blk, 0, stream>>>(ei, e1, den1, h1, acc1, E, NN, (int)H1, (int)C1);

    bias_elu<<<cdiv(NN * F1, 256), blk, 0, stream>>>(acc1, b1, NN * F1, (int)F1);

    // ===== Layer 2 =====
    dim3 g2(cdiv(NN, 128), (unsigned)(C2 / 64));
    wmma_gemm_bf16<<<g2, blk, 0, stream>>>(acc1, W2, h2, (int)NN, (int)C2, (int)F1);

    att_logits<<<cdiv(NN, 256), blk, 0, stream>>>(h2, as2, ad2, als2, ald2, NN, 1, (int)C2);

    fill_u32<<<cdiv(NN, 256), blk, 0, stream>>>(m2, ORD_NEG_INF, NN);
    fill_f32<<<cdiv(NN, 256), blk, 0, stream>>>(den2, 0.f, NN);
    fill_f32<<<cdiv(NN * C2, 256), blk, 0, stream>>>(acc2, 0.f, NN * C2);

    edge_max    <<<cdiv(TOT, 256), blk, 0, stream>>>(ei, als2, ald2, e2, m2, E, NN, 1);
    edge_exp    <<<cdiv(TOT, 256), blk, 0, stream>>>(ei, e2, m2, den2, E, NN, 1);
    edge_scatter<<<cdiv(TOT, 8),   blk, 0, stream>>>(ei, e2, den2, h2, acc2, E, NN, 1, (int)C2);

    bias_logsoftmax<<<cdiv(NN, 8), blk, 0, stream>>>(acc2, b2, out, NN);
}
